// PointTransformerLayer_42520176231038
// MI455X (gfx1250) — matrix operand order repaired and checked by reading, not compiled, not run on hardware
//
#include <hip/hip_runtime.h>

typedef _Float16 v16h __attribute__((ext_vector_type(16)));
typedef _Float16 v8h  __attribute__((ext_vector_type(8)));
typedef float    v8f  __attribute__((ext_vector_type(8)));
typedef float    v4f  __attribute__((ext_vector_type(4)));

constexpr int B_  = 4;
constexpr int N_  = 4096;
constexpr int C_  = 256;
constexpr int K_  = 16;
constexpr int M_  = B_ * N_;      // 16384 points
constexpr int MK_ = M_ * K_;      // 262144 neighbor rows
constexpr float EPS_ = 1e-5f;

// ---------------- workspace layout (bytes) ----------------
constexpr size_t AL(size_t x) { return (x + 255) & ~size_t(255); }
constexpr size_t SZ_WP   = size_t(5) * 8 * 16 * 32 * 16 * sizeof(_Float16); // 5 permuted 256x256 f16 weights
constexpr size_t SZ_MAT  = size_t(M_) * C_ * sizeof(float);                 // q/k/v f32
constexpr size_t SZ_IDX  = size_t(MK_) * sizeof(int);
constexpr size_t SZ_P3   = size_t(MK_) * 3 * sizeof(float);
constexpr size_t SZ_H16  = size_t(MK_) * C_ * sizeof(_Float16);
constexpr size_t SZ_PART = size_t(512) * C_ * sizeof(float);                // one stage-1 partial plane

constexpr size_t OFF_WP   = 0;
constexpr size_t OFF_QF   = AL(OFF_WP   + SZ_WP);
constexpr size_t OFF_KF   = AL(OFF_QF   + SZ_MAT);
constexpr size_t OFF_VF   = AL(OFF_KF   + SZ_MAT);
constexpr size_t OFF_IDX  = AL(OFF_VF   + SZ_MAT);
constexpr size_t OFF_P1   = AL(OFF_IDX  + SZ_IDX);
constexpr size_t OFF_PE3  = AL(OFF_P1   + SZ_P3);
constexpr size_t OFF_HB   = AL(OFF_PE3  + SZ_P3);
constexpr size_t OFF_H2B  = AL(OFF_HB   + SZ_H16);
constexpr size_t OFF_PA   = AL(OFF_H2B  + SZ_H16);
constexpr size_t OFF_PB   = AL(OFF_PA   + SZ_PART);
constexpr size_t OFF_ST   = AL(OFF_PB   + SZ_PART);   // stats: [0..3]=dscale [4..7]=dshift [8..263]=s1 [264..519]=t1 [520..775]=s2 [776..1031]=t2

// ---------------- K1: permute 5 weights (256x256 f32) into WMMA-B f16 layout ----------------
// B fragment: lane<16 -> col n=lane, K=0..7,16..23 ; lane>=16 -> col n=lane-16, K=8..15,24..31 (per 32-K tile)
__global__ void k_prep_w(const float* __restrict__ wq, const float* __restrict__ wk,
                         const float* __restrict__ wv, const float* __restrict__ g1,
                         const float* __restrict__ g2, _Float16* __restrict__ wp) {
  int gid = blockIdx.x * blockDim.x + threadIdx.x;
  if (gid >= 5 * 8 * 16 * 32) return;
  int lane = gid & 31; int t = gid >> 5;
  int nt = t & 15; t >>= 4;
  int kt = t & 7;  int w = t >> 3;
  const float* src = (w == 0) ? wq : (w == 1) ? wk : (w == 2) ? wv : (w == 3) ? g1 : g2;
  int kbase = kt * 32 + ((lane >= 16) ? 8 : 0);
  int n = nt * 16 + (lane & 15);
  _Float16* dst = wp + (size_t)w * 65536 + ((size_t)(kt * 16 + nt) * 32 + lane) * 16;
#pragma unroll
  for (int e = 0; e < 16; ++e) dst[e] = (_Float16)src[(size_t)(kbase + e + (e >= 8 ? 8 : 0)) * C_ + n];
}

// ---------------- K2: q/k/v = feat @ W + b  (WMMA f16, f32 accum) ----------------
__global__ void k_gemm_qkv(const float* __restrict__ feat,
                           const float* __restrict__ bq, const float* __restrict__ bk,
                           const float* __restrict__ bv, const _Float16* __restrict__ wp,
                           float* __restrict__ qf, float* __restrict__ kf, float* __restrict__ vf) {
  int lane = threadIdx.x & 31, wave = threadIdx.x >> 5;
  int rowBase = blockIdx.x * 16;
  int nt = blockIdx.y * 4 + wave;
  int z = blockIdx.z;
  const float* bias = (z == 0) ? bq : (z == 1) ? bk : bv;
  float* outp = (z == 0) ? qf : (z == 1) ? kf : vf;
  const _Float16* wpz = wp + (size_t)z * 65536;
  int m = rowBase + (lane & 15);
  int abase = (lane >= 16) ? 8 : 0;
  int n = nt * 16 + (lane & 15);
  v8f acc = {};
#pragma unroll
  for (int kt = 0; kt < 8; ++kt) {
    const float* ap = feat + (size_t)m * C_ + kt * 32 + abase;
    v4f f0 = *(const v4f*)(ap);
    v4f f1 = *(const v4f*)(ap + 4);
    v4f f2 = *(const v4f*)(ap + 16);
    v4f f3 = *(const v4f*)(ap + 20);
    v16h a;
#pragma unroll
    for (int i = 0; i < 4; ++i) {
      a[i] = (_Float16)f0[i]; a[4 + i] = (_Float16)f1[i];
      a[8 + i] = (_Float16)f2[i]; a[12 + i] = (_Float16)f3[i];
    }
    v16h b = *(const v16h*)(wpz + ((size_t)(kt * 16 + nt) * 32 + lane) * 16);
    acc = __builtin_amdgcn_wmma_f32_16x16x32_f16(false, a, false, b, (short)0, acc, false, false);
  }
  float bn = bias[n];
#pragma unroll
  for (int r = 0; r < 8; ++r) {
    int mr = rowBase + r + ((lane >= 16) ? 8 : 0);
    outp[(size_t)mr * C_ + n] = acc[r] + bn;
  }
}

// ---------------- K3: brute-force KNN (top-16 smallest sq-dist), xyz staged in LDS ----------------
__global__ void k_knn(const float* __restrict__ xyz, int* __restrict__ idx) {
  __shared__ float sx[N_], sy[N_], sz[N_];
  int b = blockIdx.y;
  const float* base = xyz + (size_t)b * N_ * 3;
  for (int i = threadIdx.x; i < N_; i += blockDim.x) {
    sx[i] = base[i * 3 + 0]; sy[i] = base[i * 3 + 1]; sz[i] = base[i * 3 + 2];
  }
  __syncthreads();
  int p = blockIdx.x * blockDim.x + threadIdx.x;
  float px = sx[p], py = sy[p], pz = sz[p];
  float bd[16]; int bi[16];
#pragma unroll
  for (int j = 0; j < 16; ++j) { bd[j] = 3.4e38f; bi[j] = 0; }
  for (int mm = 0; mm < N_; ++mm) {
    float dx = px - sx[mm], dy = py - sy[mm], dz = pz - sz[mm];
    float d = dx * dx + dy * dy + dz * dz;
    if (d < bd[15]) {
      bd[15] = d; bi[15] = mm;
#pragma unroll
      for (int j = 15; j > 0; --j) {
        if (bd[j] < bd[j - 1]) {
          float td = bd[j]; bd[j] = bd[j - 1]; bd[j - 1] = td;
          int ti = bi[j]; bi[j] = bi[j - 1]; bi[j - 1] = ti;
        }
      }
    }
  }
  int* op = idx + (size_t)(b * N_ + p) * K_;
#pragma unroll
  for (int j = 0; j < 16; ++j) op[j] = bi[j];
}

// ---------------- K4: p1 = (xyz - knn_xyz) @ d1_w + d1_b ----------------
__global__ void k_pe1(const float* __restrict__ xyz, const int* __restrict__ idx,
                      const float* __restrict__ d1w, const float* __restrict__ d1b,
                      float* __restrict__ p1) {
  int row = blockIdx.x * blockDim.x + threadIdx.x;
  int pn = row >> 4;
  int b = pn >> 12;
  int j = idx[row];
  const float* a = xyz + (size_t)pn * 3;
  const float* c = xyz + (size_t)(b * N_ + j) * 3;
  float r0 = a[0] - c[0], r1 = a[1] - c[1], r2 = a[2] - c[2];
#pragma unroll
  for (int q = 0; q < 3; ++q)
    p1[(size_t)row * 3 + q] = r0 * d1w[q] + r1 * d1w[3 + q] + r2 * d1w[6 + q] + d1b[q];
}

// ---------------- K5/K6: BN stats for 3-channel p1 (fixed-order, deterministic) ----------------
__global__ void k_stats3(const float* __restrict__ p1, float* __restrict__ part) {
  __shared__ float ss[192], sq[192];
  int tid = threadIdx.x;          // 192
  int c = tid % 3, j = tid / 3;   // j < 64
  float s = 0.f, q = 0.f;
  for (int i = 0; i < 64; ++i) {
    int r = blockIdx.x * 64 + j + i * 4096;
    float x = p1[(size_t)r * 3 + c];
    s += x; q += x * x;
  }
  ss[tid] = s; sq[tid] = q;
  __syncthreads();
  if (j == 0) {
    float S = 0.f, Q = 0.f;
    for (int t = 0; t < 64; ++t) { S += ss[c + 3 * t]; Q += sq[c + 3 * t]; }
    part[blockIdx.x * 8 + c] = S;
    part[blockIdx.x * 8 + 4 + c] = Q;
  }
}

__global__ void k_fin3(const float* __restrict__ part, const float* __restrict__ g,
                       const float* __restrict__ bb, float* __restrict__ scale,
                       float* __restrict__ shift) {
  int c = threadIdx.x;
  if (c >= 3) return;
  float S = 0.f, Q = 0.f;
  for (int p = 0; p < 64; ++p) { S += part[p * 8 + c]; Q += part[p * 8 + 4 + c]; }
  float inv = 1.0f / (float)MK_;
  float mean = S * inv;
  float var = Q * inv - mean * mean;
  float sc = rsqrtf(var + EPS_) * g[c];
  scale[c] = sc; shift[c] = bb[c] - mean * sc;
}

// ---------------- K7: pe3 = relu(bn_d(p1)); h = q - k[idx] + pe3@d2_w + d2_b -> f16 ----------------
__global__ void k_hpass(const float* __restrict__ qf, const float* __restrict__ kf,
                        const int* __restrict__ idx, const float* __restrict__ p1,
                        const float* __restrict__ dsc, const float* __restrict__ dsh,
                        const float* __restrict__ d2w, const float* __restrict__ d2b,
                        float* __restrict__ pe3, _Float16* __restrict__ hB) {
  int row = blockIdx.x;
  int c = threadIdx.x;
  int pn = row >> 4;
  int b = pn >> 12;
  int j = idx[row];
  float a0 = p1[(size_t)row * 3 + 0] * dsc[0] + dsh[0]; a0 = a0 > 0.f ? a0 : 0.f;
  float a1 = p1[(size_t)row * 3 + 1] * dsc[1] + dsh[1]; a1 = a1 > 0.f ? a1 : 0.f;
  float a2 = p1[(size_t)row * 3 + 2] * dsc[2] + dsh[2]; a2 = a2 > 0.f ? a2 : 0.f;
  if (c < 3) pe3[(size_t)row * 3 + c] = (c == 0) ? a0 : (c == 1) ? a1 : a2;
  float h = qf[(size_t)pn * C_ + c] - kf[(size_t)(b * N_ + j) * C_ + c]
          + a0 * d2w[c] + a1 * d2w[C_ + c] + a2 * d2w[2 * C_ + c] + d2b[c];
  hB[(size_t)row * C_ + c] = (_Float16)h;
}

// ---------------- K8: BN stats over an f16 [MK x 256] matrix (stage 1) ----------------
__global__ void k_stats256(const _Float16* __restrict__ x, float* __restrict__ pA,
                           float* __restrict__ pB) {
  int c = threadIdx.x;
  float s = 0.f, q = 0.f;
  for (int i = 0; i < MK_ / 512; ++i) {
    int r = blockIdx.x + i * 512;
    float v = (float)x[(size_t)r * C_ + c];
    s += v; q += v * v;
  }
  pA[(size_t)blockIdx.x * C_ + c] = s;
  pB[(size_t)blockIdx.x * C_ + c] = q;
}

__global__ void k_fin256(const float* __restrict__ pA, const float* __restrict__ pB,
                         const float* __restrict__ g, const float* __restrict__ bb,
                         float* __restrict__ scale, float* __restrict__ shift) {
  int c = threadIdx.x;
  float S = 0.f, Q = 0.f;
  for (int p = 0; p < 512; ++p) { S += pA[(size_t)p * C_ + c]; Q += pB[(size_t)p * C_ + c]; }
  float inv = 1.0f / (float)MK_;
  float mean = S * inv;
  float var = Q * inv - mean * mean;
  float sc = rsqrtf(var + EPS_) * g[c];
  scale[c] = sc; shift[c] = bb[c] - mean * sc;
}

// ---------------- K10: h2 = relu(bn1(h)) @ g1_w + g1_b  (affine fused into A load) ----------------
__global__ void k_gemm_h1(const _Float16* __restrict__ hB, const _Float16* __restrict__ wp,
                          const float* __restrict__ s1, const float* __restrict__ t1,
                          const float* __restrict__ g1b, _Float16* __restrict__ h2B) {
  __shared__ float ls[C_], lt[C_];
  int tid = threadIdx.x;  // 128
  ls[tid] = s1[tid]; lt[tid] = t1[tid];
  ls[tid + 128] = s1[tid + 128]; lt[tid + 128] = t1[tid + 128];
  __syncthreads();
  int lane = tid & 31, wave = tid >> 5;
  int rowBase = blockIdx.x * 16;
  int nt = blockIdx.y * 4 + wave;
  int m = rowBase + (lane & 15);
  int abase = (lane >= 16) ? 8 : 0;
  int n = nt * 16 + (lane & 15);
  const _Float16* g1p = wp + (size_t)3 * 65536;
  v8f acc = {};
#pragma unroll
  for (int kt = 0; kt < 8; ++kt) {
    int k0 = kt * 32 + abase;
    const _Float16* ap = hB + (size_t)m * C_ + k0;
    v8h u0 = *(const v8h*)(ap);
    v8h u1 = *(const v8h*)(ap + 16);
    v16h a;
#pragma unroll
    for (int e = 0; e < 8; ++e) {
      float f0 = (float)u0[e] * ls[k0 + e] + lt[k0 + e];
      float f1 = (float)u1[e] * ls[k0 + 16 + e] + lt[k0 + 16 + e];
      a[e]     = (_Float16)(f0 > 0.f ? f0 : 0.f);
      a[8 + e] = (_Float16)(f1 > 0.f ? f1 : 0.f);
    }
    v16h b = *(const v16h*)(g1p + ((size_t)(kt * 16 + nt) * 32 + lane) * 16);
    acc = __builtin_amdgcn_wmma_f32_16x16x32_f16(false, a, false, b, (short)0, acc, false, false);
  }
  float bn = g1b[n];
#pragma unroll
  for (int r = 0; r < 8; ++r) {
    int mr = rowBase + r + ((lane >= 16) ? 8 : 0);
    h2B[(size_t)mr * C_ + n] = (_Float16)(acc[r] + bn);
  }
}

// ---------------- K13: attn = relu(bn2(h2)) @ g2_w + g2_b ; softmax over K ; out = sum_k (v+pe)*attn
// One block = one point: 16 rows (=K neighbors) x 256 cols; A tile staged in LDS, 16 waves x 16-col tiles.
__global__ void k_attn_out(const _Float16* __restrict__ h2B, const _Float16* __restrict__ wp,
                           const float* __restrict__ s2, const float* __restrict__ t2,
                           const float* __restrict__ g2b, const int* __restrict__ idx,
                           const float* __restrict__ pe3, const float* __restrict__ vf,
                           const float* __restrict__ d2w, const float* __restrict__ d2b,
                           float* __restrict__ out) {
  __shared__ _Float16 aT[16 * C_];
  __shared__ int sidx[16];
  __shared__ float spe[48];
  int tid = threadIdx.x;  // 512
  int p = blockIdx.x;
  int b = p >> 12;
#pragma unroll
  for (int i = 0; i < 8; ++i) {
    int e = tid + i * 512;
    int r = e >> 8, c = e & 255;
    float v = (float)h2B[((size_t)p * 16 + r) * C_ + c] * s2[c] + t2[c];
    aT[r * C_ + c] = (_Float16)(v > 0.f ? v : 0.f);
  }
  if (tid < 16) sidx[tid] = idx[(size_t)p * 16 + tid];
  if (tid >= 16 && tid < 64) spe[tid - 16] = pe3[(size_t)p * 48 + (tid - 16)];
  __syncthreads();
  int lane = tid & 31, wave = tid >> 5;
  int nt = wave;
  int n = nt * 16 + (lane & 15);
  int abase = (lane >= 16) ? 8 : 0;
  int arow = lane & 15;
  const _Float16* g2p = wp + (size_t)4 * 65536;
  v8f acc = {};
#pragma unroll
  for (int kt = 0; kt < 8; ++kt) {
    const _Float16* ap = &aT[arow * C_ + kt * 32 + abase];
    v8h u0 = *(const v8h*)ap;
    v8h u1 = *(const v8h*)(ap + 16);
    v16h a;
#pragma unroll
    for (int e = 0; e < 8; ++e) { a[e] = u0[e]; a[8 + e] = u1[e]; }
    v16h bf = *(const v16h*)(g2p + ((size_t)(kt * 16 + nt) * 32 + lane) * 16);
    acc = __builtin_amdgcn_wmma_f32_16x16x32_f16(false, a, false, bf, (short)0, acc, false, false);
  }
  float bias = g2b[n];
  float vals[8];
  float mx = -3.4e38f;
#pragma unroll
  for (int r = 0; r < 8; ++r) { vals[r] = acc[r] + bias; mx = fmaxf(mx, vals[r]); }
  mx = fmaxf(mx, __shfl_xor(mx, 16));          // combine the two 8-row halves (lanes l <-> l^16)
  float sum = 0.f;
#pragma unroll
  for (int r = 0; r < 8; ++r) { vals[r] = __expf(vals[r] - mx); sum += vals[r]; }
  sum += __shfl_xor(sum, 16);
  float inv = 1.0f / sum;
  int khalf = (lane >= 16) ? 8 : 0;
  float partial = 0.f;
#pragma unroll
  for (int r = 0; r < 8; ++r) {
    int k = khalf + r;
    int j = sidx[k];
    float vv = vf[((size_t)(b * N_) + j) * C_ + n];
    float pe = spe[k * 3 + 0] * d2w[n] + spe[k * 3 + 1] * d2w[C_ + n]
             + spe[k * 3 + 2] * d2w[2 * C_ + n] + d2b[n];
    partial += (vv + pe) * (vals[r] * inv);
  }
  partial += __shfl_xor(partial, 16);
  if (lane < 16) out[(size_t)p * C_ + n] = partial;
}

// ---------------- launch ----------------
extern "C" void kernel_launch(void* const* d_in, const int* in_sizes, int n_in,
                              void* d_out, int out_size, void* d_ws, size_t ws_size,
                              hipStream_t stream) {
  (void)in_sizes; (void)n_in; (void)out_size; (void)ws_size;
  const float* xyz  = (const float*)d_in[0];
  const float* feat = (const float*)d_in[1];
  const float* wq   = (const float*)d_in[2];
  const float* bq   = (const float*)d_in[3];
  const float* wk   = (const float*)d_in[4];
  const float* bk   = (const float*)d_in[5];
  const float* wv   = (const float*)d_in[6];
  const float* bv   = (const float*)d_in[7];
  const float* d1w  = (const float*)d_in[8];
  const float* d1b  = (const float*)d_in[9];
  const float* bndg = (const float*)d_in[10];
  const float* bndb = (const float*)d_in[11];
  const float* d2w  = (const float*)d_in[12];
  const float* d2b  = (const float*)d_in[13];
  const float* g1g  = (const float*)d_in[14];
  const float* g1bb = (const float*)d_in[15];
  const float* g1w  = (const float*)d_in[16];
  const float* g1b  = (const float*)d_in[17];
  const float* g2g  = (const float*)d_in[18];
  const float* g2bb = (const float*)d_in[19];
  const float* g2w  = (const float*)d_in[20];
  const float* g2b  = (const float*)d_in[21];

  char* ws = (char*)d_ws;
  _Float16* WP  = (_Float16*)(ws + OFF_WP);
  float*    QF  = (float*)(ws + OFF_QF);
  float*    KF  = (float*)(ws + OFF_KF);
  float*    VF  = (float*)(ws + OFF_VF);
  int*      IDX = (int*)(ws + OFF_IDX);
  float*    P1  = (float*)(ws + OFF_P1);
  float*    PE3 = (float*)(ws + OFF_PE3);
  _Float16* HB  = (_Float16*)(ws + OFF_HB);
  _Float16* H2B = (_Float16*)(ws + OFF_H2B);
  float*    PA  = (float*)(ws + OFF_PA);
  float*    PB  = (float*)(ws + OFF_PB);
  float*    ST  = (float*)(ws + OFF_ST);
  float *dsc = ST, *dsh = ST + 4;
  float *s1 = ST + 8, *t1 = ST + 8 + 256;
  float *s2 = ST + 8 + 512, *t2 = ST + 8 + 768;

  k_prep_w<<<80, 256, 0, stream>>>(wq, wk, wv, g1w, g2w, WP);
  k_gemm_qkv<<<dim3(M_ / 16, 4, 3), 128, 0, stream>>>(feat, bq, bk, bv, WP, QF, KF, VF);
  k_knn<<<dim3(16, 4), 256, 0, stream>>>(xyz, IDX);
  k_pe1<<<MK_ / 256, 256, 0, stream>>>(xyz, IDX, d1w, d1b, P1);
  k_stats3<<<64, 192, 0, stream>>>(P1, PA);
  k_fin3<<<1, 32, 0, stream>>>(PA, bndg, bndb, dsc, dsh);
  k_hpass<<<MK_, 256, 0, stream>>>(QF, KF, IDX, P1, dsc, dsh, d2w, d2b, PE3, HB);
  k_stats256<<<512, 256, 0, stream>>>(HB, PA, PB);
  k_fin256<<<1, 256, 0, stream>>>(PA, PB, g1g, g1bb, s1, t1);
  k_gemm_h1<<<dim3(MK_ / 16, 4), 128, 0, stream>>>(HB, WP, s1, t1, g1b, H2B);
  k_stats256<<<512, 256, 0, stream>>>(H2B, PA, PB);
  k_fin256<<<1, 256, 0, stream>>>(PA, PB, g2g, g2bb, s2, t2);
  k_attn_out<<<M_, 512, 0, stream>>>(H2B, WP, s2, t2, g2b, IDX, PE3, VF, d2w, d2b, (float*)d_out);
}
